// GNP_50852412785142
// MI455X (gfx1250) — compile-verified, matrix operand order checked by reading, not run on hardware
//
#include <hip/hip_runtime.h>
#include <hip/hip_bf16.h>
#include <stdint.h>

#define NN 50000
#define NE 800000
#define ND 16
#define ED 16
#define H  128
#define OD 4
#define NB 3
#define KMSG 160   // 144 padded up to a multiple of 32

typedef __attribute__((ext_vector_type(16))) __bf16 v16bf;
typedef __attribute__((ext_vector_type(8)))  float  v8f;
typedef __attribute__((ext_vector_type(4)))  int    v4i;

union Frag16 { v16bf v; __bf16 e[16]; uint32_t u[8]; uint4 q[2]; };

// native f32 -> bf16 convert (let the backend pick the hw convert)
static __device__ __forceinline__ __bf16 f2bf(float f) { return (__bf16)f; }

// K position (within a 32-wide K chunk) of VGPR v's first bf16, per ISA
// 16-bit A-matrix layout: VGPR0-3 hold K pairs {0..15}, VGPR4-7 hold {16..31},
// lanes 16-31 shifted by 8.
static __device__ __forceinline__ int kmap(int v, int kh) {
  return (v < 4) ? (2 * v + 8 * kh) : (16 + 2 * (v - 4) + 8 * kh);
}

static __device__ __forceinline__ int clampN(int i) {
  return i < 0 ? 0 : (i >= NN ? NN - 1 : i);
}

// Load one pre-packed 16x16(x32) weight fragment: 512 bf16 in lane order,
// lane's 16 elements contiguous -> two b128 loads.
static __device__ __forceinline__ v16bf load_frag(const __bf16* __restrict__ base, int lane) {
  Frag16 f;
  const uint4* p = (const uint4*)base + lane * 2;
  f.q[0] = p[0];
  f.q[1] = p[1];
  return f.v;
}

// Async global -> LDS 16-byte copy (CDNA5 ASYNCcnt path), with asm fallback.
static __device__ __forceinline__ void async_b128(const void* gaddr, void* lds) {
#if __has_builtin(__builtin_amdgcn_global_load_async_to_lds_b128)
  __builtin_amdgcn_global_load_async_to_lds_b128(
      (__attribute__((address_space(1))) v4i*)gaddr,
      (__attribute__((address_space(3))) v4i*)lds, 0, 0);
#else
  unsigned l = (unsigned)(uintptr_t)lds;
  unsigned long long g = (unsigned long long)(uintptr_t)gaddr;
  asm volatile("global_load_async_to_lds_b128 %0, %1, off" :: "v"(l), "v"(g) : "memory");
#endif
}

static __device__ __forceinline__ void wait_async0() {
#if __has_builtin(__builtin_amdgcn_s_wait_asynccnt)
  __builtin_amdgcn_s_wait_asynccnt(0);
#else
  asm volatile("s_wait_asynccnt 0" ::: "memory");
#endif
}

// ---------------- conversion / packing kernels ----------------

__global__ void k_f32_to_bf16(const float* __restrict__ s, __bf16* __restrict__ d, int n) {
  for (int i = blockIdx.x * blockDim.x + threadIdx.x; i < n; i += gridDim.x * blockDim.x)
    d[i] = f2bf(s[i]);
}

// Pack [nblocks][ksrc][H] f32 weights into WMMA-fragment lane order:
// layout (((b*8 + ctile)*kt_cnt + kt)*32 + lane)*16 + elem, zero-padded K.
__global__ void k_pack_w(const float* __restrict__ s, __bf16* __restrict__ d,
                         int ksrc, int kt_cnt, int nblocks) {
  int total = nblocks * 8 * kt_cnt * 512;
  for (int i = blockIdx.x * blockDim.x + threadIdx.x; i < total; i += gridDim.x * blockDim.x) {
    int e    = i & 15;
    int lane = (i >> 4) & 31;
    int fi   = i >> 9;                 // fragment index
    int kt   = fi % kt_cnt;
    int c    = (fi / kt_cnt) & 7;
    int b    = fi / (kt_cnt * 8);
    int kh = lane >> 4, ln = lane & 15;
    int k = kt * 32 + kmap(e >> 1, kh) + (e & 1);
    int n = c * 16 + ln;
    float val = (k < ksrc) ? s[((size_t)b * ksrc + k) * H + n] : 0.0f;
    d[i] = f2bf(val);
  }
}

__global__ void k_zero(float* __restrict__ p, int n) {
  for (int i = blockIdx.x * blockDim.x + threadIdx.x; i < n; i += gridDim.x * blockDim.x)
    p[i] = 0.0f;
}

// ---------------- lift: h = x @ lift_w + lift_b ----------------
// one 16x16 output tile per wave; K = 16 padded to 32 -> single WMMA

__global__ void k_lift(const float* __restrict__ x, const __bf16* __restrict__ lwp,
                       const float* __restrict__ lb, float* __restrict__ hf,
                       __bf16* __restrict__ hb) {
  int wave = threadIdx.x >> 5;   // 8 column tiles
  int lane = threadIdx.x & 31;
  int kh = lane >> 4, ln = lane & 15;
  int m0 = blockIdx.x * 16, n0 = wave * 16;
  int row = m0 + ln;

  Frag16 a;
#pragma unroll
  for (int v = 0; v < 8; v++) {
    int k = kmap(v, kh);
    a.e[2 * v]     = (k     < ND) ? f2bf(x[row * ND + k])     : f2bf(0.0f);
    a.e[2 * v + 1] = (k + 1 < ND) ? f2bf(x[row * ND + k + 1]) : f2bf(0.0f);
  }
  v16bf b = load_frag(lwp + (size_t)wave * 512, lane);
  v8f c = {};
  c = __builtin_amdgcn_wmma_f32_16x16x32_bf16(false, a.v, false, b, (short)0, c, false, false);
#pragma unroll
  for (int r = 0; r < 8; r++) {
    int m = m0 + r + 8 * kh, n = n0 + ln;
    float val = c[r] + lb[n];
    hf[(size_t)m * H + n] = val;
    hb[(size_t)m * H + n] = f2bf(val);
  }
}

// ---------------- edge messages + scatter-add ----------------
// Persistent workgroups; msg weight fragments resident in registers.
// Per 16-edge tile: async-stage [h[src] || e_ij] (16 x 160 bf16) into LDS,
// 5 chained WMMAs, scatter 16x16 f32 results with global atomics into agg.

__global__ void k_msg(const __bf16* __restrict__ hb, const __bf16* __restrict__ eab,
                      const __bf16* __restrict__ mwp,  // packed [8 ctiles][5 kt] frags
                      const float* __restrict__ mb,    // [H]
                      const int* __restrict__ src, const int* __restrict__ dst,
                      float* __restrict__ agg) {
  __shared__ uint4 smq[16 * 20];   // 16 rows x 160 bf16 (20 quads; quads 18,19 = 0)
  __shared__ int sj[16];
  __shared__ int sd[16];

  int tid = threadIdx.x;
  int wave = tid >> 5, lane = tid & 31, kh = lane >> 4, ln = lane & 15;
  int n0 = wave * 16;

  // zero the constant K-padding region once (k = 144..159)
  if (tid < 16) {
    uint4 z; z.x = z.y = z.z = z.w = 0u;
    smq[tid * 20 + 18] = z;
    smq[tid * 20 + 19] = z;
  }

  // B fragments: loaded once, reused for every edge tile
  v16bf B[5];
#pragma unroll
  for (int kt = 0; kt < 5; kt++)
    B[kt] = load_frag(mwp + ((size_t)wave * 5 + kt) * 512, lane);
  float bias = mb[n0 + ln];

  const uint32_t* sm32 = (const uint32_t*)smq;
  const int ntiles = NE / 16;
  for (int t = blockIdx.x; t < ntiles; t += gridDim.x) {
    __syncthreads();   // LDS reuse guard (also covers the one-time zero fill)
    if (tid < 16) {
      sj[tid] = clampN(src[t * 16 + tid]);
      sd[tid] = clampN(dst[t * 16 + tid]);
    }
    __syncthreads();
    // stage m_in tile: 16 rows x 18 quads (h[src] = 16 quads, edge_attr = 2)
    for (int i = tid; i < 16 * 18; i += blockDim.x) {
      int row = i / 18, q = i % 18;
      const __bf16* g = (q < 16) ? (hb + (size_t)sj[row] * H + q * 8)
                                 : (eab + (size_t)(t * 16 + row) * ED + (q - 16) * 8);
      async_b128(g, &smq[row * 20 + q]);
    }
    wait_async0();
    __syncthreads();

    v8f c = {};
#pragma unroll
    for (int kt = 0; kt < 5; kt++) {
      Frag16 a;
#pragma unroll
      for (int v = 0; v < 8; v++)
        a.u[v] = sm32[ln * 80 + kt * 16 + kmap(v, kh) / 2];
      c = __builtin_amdgcn_wmma_f32_16x16x32_bf16(false, a.v, false, B[kt], (short)0, c,
                                                  false, false);
    }
#pragma unroll
    for (int r = 0; r < 8; r++) {
      int m = r + 8 * kh;               // edge row within tile
      int n = n0 + ln;
      atomicAdd(&agg[(size_t)sd[m] * H + n], c[r] + bias);
    }
  }
}

// ---------------- node update: out = h@root_w + root_b + agg + h (+gelu) ----

__global__ void k_node(const __bf16* __restrict__ hbi, const float* __restrict__ hfi,
                       const float* __restrict__ agg, const __bf16* __restrict__ rwp,
                       const float* __restrict__ rb, float* __restrict__ hfo,
                       __bf16* __restrict__ hbo, int use_act) {
  __shared__ uint4 snq[16 * 16];   // 16 rows x 128 bf16
  int tid = threadIdx.x, wave = tid >> 5, lane = tid & 31, kh = lane >> 4, ln = lane & 15;
  int m0 = blockIdx.x * 16, n0 = wave * 16;

  {   // async-stage the 16x128 bf16 activation tile: exactly 1 quad per thread
    int row = tid >> 4, q = tid & 15;
    async_b128(hbi + (size_t)(m0 + row) * H + q * 8, &snq[tid]);
    wait_async0();
  }
  __syncthreads();

  const uint32_t* sm32 = (const uint32_t*)snq;
  v8f c = {};
#pragma unroll
  for (int kt = 0; kt < 4; kt++) {
    Frag16 a;
#pragma unroll
    for (int v = 0; v < 8; v++)
      a.u[v] = sm32[ln * 64 + kt * 16 + kmap(v, kh) / 2];
    v16bf b = load_frag(rwp + ((size_t)wave * 4 + kt) * 512, lane);
    c = __builtin_amdgcn_wmma_f32_16x16x32_bf16(false, a.v, false, b, (short)0, c, false, false);
  }
#pragma unroll
  for (int r = 0; r < 8; r++) {
    int m = m0 + r + 8 * kh, n = n0 + ln;
    float val = c[r] + rb[n] + agg[(size_t)m * H + n] + hfi[(size_t)m * H + n];
    if (use_act) {   // tanh-approx gelu (jax.nn.gelu default)
      float t = tanhf(0.7978845608028654f * (val + 0.044715f * val * val * val));
      val = 0.5f * val * (1.0f + t);
    }
    hfo[(size_t)m * H + n] = val;
    hbo[(size_t)m * H + n] = f2bf(val);
  }
}

// ---------------- projection: [N,128] @ [128,4] + b ----------------

__global__ void k_proj(const float* __restrict__ hf, const float* __restrict__ pw,
                       const float* __restrict__ pb, float* __restrict__ out) {
  int node = blockIdx.x * blockDim.x + threadIdx.x;
  if (node >= NN) return;
  float acc[OD] = {pb[0], pb[1], pb[2], pb[3]};
  const float* hr = hf + (size_t)node * H;
  for (int k = 0; k < H; k++) {
    float hv = hr[k];
#pragma unroll
    for (int o = 0; o < OD; o++) acc[o] += hv * pw[k * OD + o];
  }
#pragma unroll
  for (int o = 0; o < OD; o++) out[(size_t)node * OD + o] = acc[o];
}

// ---------------- launch ----------------

extern "C" void kernel_launch(void* const* d_in, const int* in_sizes, int n_in,
                              void* d_out, int out_size, void* d_ws, size_t ws_size,
                              hipStream_t stream) {
  (void)in_sizes; (void)n_in; (void)out_size; (void)ws_size;

  const float* x  = (const float*)d_in[0];
  const int*   ei = (const int*)d_in[1];      // edge_index [2, NE]
  const float* ea = (const float*)d_in[2];
  const float* lw = (const float*)d_in[3];
  const float* lb = (const float*)d_in[4];
  const float* rw = (const float*)d_in[5];
  const float* rb = (const float*)d_in[6];
  const float* mw = (const float*)d_in[7];
  const float* mb = (const float*)d_in[8];
  const float* pw = (const float*)d_in[9];
  const float* pb = (const float*)d_in[10];
  float* out = (float*)d_out;
  const int* src = ei;
  const int* dst = ei + NE;

  char* ws = (char*)d_ws;
  size_t off = 0;
  auto alloc = [&](size_t bytes) -> char* {
    char* p = ws + off;
    off += (bytes + 1023) & ~(size_t)1023;
    return p;
  };
  float*  hf0 = (float*) alloc((size_t)NN * H * 4);
  float*  hf1 = (float*) alloc((size_t)NN * H * 4);
  __bf16* hb0 = (__bf16*)alloc((size_t)NN * H * 2);
  __bf16* hb1 = (__bf16*)alloc((size_t)NN * H * 2);
  float*  agg = (float*) alloc((size_t)NN * H * 4);
  __bf16* eab = (__bf16*)alloc((size_t)NE * ED * 2);
  __bf16* mwp = (__bf16*)alloc((size_t)NB * 8 * 5 * 512 * 2);  // packed msg weights
  __bf16* rwp = (__bf16*)alloc((size_t)NB * 8 * 4 * 512 * 2);  // packed root weights
  __bf16* lwp = (__bf16*)alloc((size_t)8 * 512 * 2);           // packed lift weights

  k_f32_to_bf16<<<512, 256, 0, stream>>>(ea, eab, NE * ED);
  k_pack_w<<<64, 256, 0, stream>>>(mw, mwp, H + ED, 5, NB);
  k_pack_w<<<64, 256, 0, stream>>>(rw, rwp, H, 4, NB);
  k_pack_w<<<16, 256, 0, stream>>>(lw, lwp, ND, 1, 1);

  k_lift<<<NN / 16, 256, 0, stream>>>(x, lwp, lb, hf0, hb0);

  float* hfi = hf0; __bf16* hbi = hb0;
  float* hfo = hf1; __bf16* hbo = hb1;
  for (int blk = 0; blk < NB; blk++) {
    k_zero<<<512, 256, 0, stream>>>(agg, NN * H);
    k_msg<<<1024, 256, 0, stream>>>(hbi, eab, mwp + (size_t)blk * 8 * 5 * 512, mb + blk * H,
                                    src, dst, agg);
    k_node<<<NN / 16, 256, 0, stream>>>(hbi, hfi, agg, rwp + (size_t)blk * 8 * 4 * 512,
                                        rb + blk * H, hfo, hbo, (blk < NB - 1) ? 1 : 0);
    float* tf = hfi; hfi = hfo; hfo = tf;
    __bf16* tb = hbi; hbi = hbo; hbo = tb;
  }
  k_proj<<<(NN + 255) / 256, 256, 0, stream>>>(hfi, pw, pb, out);
}
